// DevignModel_4346506904211
// MI455X (gfx1250) — matrix operand order repaired and checked by reading, not compiled, not run on hardware
//
#include <hip/hip_runtime.h>
#include <hip/hip_bf16.h>

// ---------------- model dims ----------------
#define BATCH 64
#define SEQN  512
#define DINF  106
#define OUTF  256
#define NTYPE 6
#define NEDGE 524288
#define NSTEP 6
#define CATF  362
#define BN    (BATCH*SEQN)   // 32768
#define ZP    384            // Z-tower N and K padded to /32
#define KMAX  384
#define LDSPAD 8             // bf16 elements of row padding (16B) vs 64 banks

typedef __bf16 bf16_t;
typedef bf16_t v16bf __attribute__((ext_vector_type(16)));
typedef bf16_t v8bf  __attribute__((ext_vector_type(8)));
typedef float  v8f   __attribute__((ext_vector_type(8)));

__device__ __forceinline__ v8bf load8bf(const bf16_t* p) {
  return *reinterpret_cast<const v8bf*>(p);   // 16B -> global_load_b128 / ds_load_b128
}
__device__ __forceinline__ v16bf cat16(v8bf lo, v8bf hi) {
  v16bf r;
#pragma unroll
  for (int i = 0; i < 8; ++i) { r[i] = lo[i]; r[i + 8] = hi[i]; }
  return r;
}

// =====================================================================
// WMMA GEMM:  C[M,N](f32) = act( A[M,K](bf16,row) @ B[N,K](bf16,row)^T + bias )
// Block = 4 waves = 128x32 output panel. The 32xK weight panel is staged
// once per block into LDS with CDNA5 async copies
// (global_load_async_to_lds_b128, ASYNCcnt, s_wait_asynccnt), then each
// wave computes a 32x32 register tile: per 32-k chunk 4x global b128 (A,
// double-buffered) + 4x ds_load_b128 (B) feed 4x v_wmma_f32_16x16x32_bf16.
// Conv mode (rows_per_out>0): output row m -> b=m/rpo, l=m%rpo,
//   A row = b*rpi + l + shift, accumulated over nshift shifts
//   (K=3 VALID conv1d over channels-last data, weights as [nshift][N][ldb]).
// Requires: M%128==0, N%32==0, K%32==0, K<=KMAX.
// =====================================================================
__global__ __launch_bounds__(128)
void wmma_gemm_bf16(const bf16_t* __restrict__ A, int lda,
                    const bf16_t* __restrict__ Bw, int ldb,
                    const float* __restrict__ bias,
                    float* __restrict__ C, int ldc,
                    int M, int N, int K,
                    int rows_per_out, int rows_per_in, int nshift, int relu)
{
  __shared__ bf16_t Bs[32 * (KMAX + LDSPAD)];
  const int wave = threadIdx.x >> 5;
  const int lane = threadIdx.x & 31;
  const int tilesN = N >> 5;
  const int tM = blockIdx.x / tilesN;          // 128-row block index
  const int tN = blockIdx.x - tM * tilesN;     // 32-col block index
  const int half = lane >> 4;
  const int l16  = lane & 15;
  const int mA0 = tM * 128 + wave * 32 + l16;  // A rows this lane streams
  const int mA1 = mA0 + 16;
  const int nB0 = tN * 32 + l16;               // output columns
  const int nB1 = nB0 + 16;
  const int ldsK = K + LDSPAD;
  const int kv = K >> 3;                       // 16B chunks per B row

  v8f acc00 = {}, acc01 = {}, acc10 = {}, acc11 = {};

  for (int s = 0; s < nshift; ++s) {
    // ---- cooperative async stage of B panel [32 x K] into LDS ----
    if (s) __syncthreads();                    // previous shift done reading
    {
      const bf16_t* Bpanel = Bw + (size_t)s * (size_t)N * ldb
                                + (size_t)(tN * 32) * ldb;
      for (int c = threadIdx.x; c < 32 * kv; c += 128) {
        int row = c / kv;
        int ko  = (c - row * kv) << 3;
        unsigned long long g =
            (unsigned long long)(uintptr_t)(Bpanel + (size_t)row * ldb + ko);
        unsigned l = (unsigned)(uintptr_t)(&Bs[row * ldsK + ko]);
        asm volatile("global_load_async_to_lds_b128 %0, %1, off"
                     :: "v"(l), "v"(g) : "memory");
      }
      asm volatile("s_wait_asynccnt 0x0" ::: "memory");
    }
    __syncthreads();

    // ---- per-wave compute over K ----
    int ar0, ar1;
    if (rows_per_out > 0) {
      int b0 = mA0 / rows_per_out, l0 = mA0 - b0 * rows_per_out;
      int b1 = mA1 / rows_per_out, l1 = mA1 - b1 * rows_per_out;
      ar0 = b0 * rows_per_in + l0 + s;
      ar1 = b1 * rows_per_in + l1 + s;
    } else { ar0 = mA0; ar1 = mA1; }
    const bf16_t* __restrict__ A0 = A + (size_t)ar0 * lda;
    const bf16_t* __restrict__ A1 = A + (size_t)ar1 * lda;
    const bf16_t* Bl0 = &Bs[l16 * ldsK];        // B frag row n = lane&15
    const bf16_t* Bl1 = &Bs[(l16 + 16) * ldsK];

    // preload A chunk k=0 (frag: k = 8*half+[0,8) and 16+8*half+[0,8))
    v8bf a00 = load8bf(A0 + 8 * half);
    v8bf a01 = load8bf(A0 + 16 + 8 * half);
    v8bf a10 = load8bf(A1 + 8 * half);
    v8bf a11 = load8bf(A1 + 16 + 8 * half);
    for (int k0 = 32; k0 < K; k0 += 32) {
      // next-chunk A loads in flight while this chunk computes
      v8bf na00 = load8bf(A0 + k0 + 8 * half);
      v8bf na01 = load8bf(A0 + k0 + 16 + 8 * half);
      v8bf na10 = load8bf(A1 + k0 + 8 * half);
      v8bf na11 = load8bf(A1 + k0 + 16 + 8 * half);
      int kc = k0 - 32;
      // B frag from LDS: k = 8*half+[0,8) and 16+8*half+[0,8)
      v8bf b00 = load8bf(Bl0 + kc + 8 * half);
      v8bf b01 = load8bf(Bl0 + kc + 16 + 8 * half);
      v8bf b10 = load8bf(Bl1 + kc + 8 * half);
      v8bf b11 = load8bf(Bl1 + kc + 16 + 8 * half);
      v16bf fa0 = cat16(a00, a01), fa1 = cat16(a10, a11);
      v16bf fb0 = cat16(b00, b01), fb1 = cat16(b10, b11);
      acc00 = __builtin_amdgcn_wmma_f32_16x16x32_bf16(false, fa0, false, fb0,
                                                      (short)0, acc00, false, false);
      acc01 = __builtin_amdgcn_wmma_f32_16x16x32_bf16(false, fa0, false, fb1,
                                                      (short)0, acc01, false, false);
      acc10 = __builtin_amdgcn_wmma_f32_16x16x32_bf16(false, fa1, false, fb0,
                                                      (short)0, acc10, false, false);
      acc11 = __builtin_amdgcn_wmma_f32_16x16x32_bf16(false, fa1, false, fb1,
                                                      (short)0, acc11, false, false);
      a00 = na00; a01 = na01; a10 = na10; a11 = na11;
    }
    { // tail chunk
      int kc = K - 32;
      v8bf b00 = load8bf(Bl0 + kc + 8 * half);
      v8bf b01 = load8bf(Bl0 + kc + 16 + 8 * half);
      v8bf b10 = load8bf(Bl1 + kc + 8 * half);
      v8bf b11 = load8bf(Bl1 + kc + 16 + 8 * half);
      v16bf fa0 = cat16(a00, a01), fa1 = cat16(a10, a11);
      v16bf fb0 = cat16(b00, b01), fb1 = cat16(b10, b11);
      acc00 = __builtin_amdgcn_wmma_f32_16x16x32_bf16(false, fa0, false, fb0,
                                                      (short)0, acc00, false, false);
      acc01 = __builtin_amdgcn_wmma_f32_16x16x32_bf16(false, fa0, false, fb1,
                                                      (short)0, acc01, false, false);
      acc10 = __builtin_amdgcn_wmma_f32_16x16x32_bf16(false, fa1, false, fb0,
                                                      (short)0, acc10, false, false);
      acc11 = __builtin_amdgcn_wmma_f32_16x16x32_bf16(false, fa1, false, fb1,
                                                      (short)0, acc11, false, false);
    }
  }
  const float bv0 = bias ? bias[nB0] : 0.0f;
  const float bv1 = bias ? bias[nB1] : 0.0f;
#pragma unroll
  for (int r = 0; r < 8; ++r) {
    int m0 = tM * 128 + wave * 32 + r + 8 * half;  // D: row = r+8*half, col = lane&15
    int m1 = m0 + 16;
    float v00 = acc00[r] + bv0, v01 = acc01[r] + bv1;
    float v10 = acc10[r] + bv0, v11 = acc11[r] + bv1;
    if (relu) {
      v00 = fmaxf(v00, 0.0f); v01 = fmaxf(v01, 0.0f);
      v10 = fmaxf(v10, 0.0f); v11 = fmaxf(v11, 0.0f);
    }
    C[(size_t)m0 * ldc + nB0] = v00;
    C[(size_t)m0 * ldc + nB1] = v01;
    C[(size_t)m1 * ldc + nB0] = v10;
    C[(size_t)m1 * ldc + nB1] = v11;
  }
}

// ---------------- elementwise / data-movement kernels ----------------
__global__ void k_pad_h(const float* __restrict__ nf, float* __restrict__ h,
                        bf16_t* __restrict__ hb, int total) {
  int i = blockIdx.x * blockDim.x + threadIdx.x;
  if (i >= total) return;
  int row = i >> 8, c = i & 255;
  float v = (c < DINF) ? nf[row * DINF + c] : 0.0f;
  h[i] = v; hb[i] = (bf16_t)v;
}

__global__ void k_f32_to_bf16(const float* __restrict__ in,
                              bf16_t* __restrict__ out, int n) {
  int i = blockIdx.x * blockDim.x + threadIdx.x;
  if (i < n) out[i] = (bf16_t)in[i];
}

__global__ void k_zero_f32(float* __restrict__ p, int n) {
  int i = blockIdx.x * blockDim.x + threadIdx.x;
  if (i < n) p[i] = 0.0f;
}

// conv weight [Co,Ci,K] f32 -> [K, Np, Kp] bf16, zero-padded
__global__ void k_repack_conv(const float* __restrict__ w, bf16_t* __restrict__ out,
                              int Co, int Ci, int Kk, int Np, int Kp) {
  int i = blockIdx.x * blockDim.x + threadIdx.x;
  int total = Kk * Np * Kp;
  if (i >= total) return;
  int k = i / (Np * Kp);
  int rem = i - k * (Np * Kp);
  int n = rem / Kp, c = rem - n * Kp;
  float v = (n < Co && c < Ci) ? w[(size_t)n * Ci * Kk + (size_t)c * Kk + k] : 0.0f;
  out[i] = (bf16_t)v;
}

__global__ void k_pad_bias(const float* __restrict__ in, float* __restrict__ out,
                           int C, int Cp) {
  int i = blockIdx.x * blockDim.x + threadIdx.x;
  if (i < Cp) out[i] = (i < C) ? in[i] : 0.0f;
}

// a[dst[e], :] += proj[src[e], :]  for edges of type t (proj already has b_e[t])
__global__ void k_scatter(const float* __restrict__ proj,
                          const int* __restrict__ src, const int* __restrict__ dst,
                          const int* __restrict__ et, int t,
                          float* __restrict__ a) {
  int e = blockIdx.x;
  if (et[e] != t) return;
  int c = threadIdx.x;                 // 256 threads = OUTF columns
  int s = src[e], d = dst[e];
  float v = proj[(size_t)s * OUTF + c];
  __hip_atomic_fetch_add(&a[(size_t)d * OUTF + c], v,
                         __ATOMIC_RELAXED, __HIP_MEMORY_SCOPE_AGENT);
}

__device__ __forceinline__ float sigf(float x) { return 1.0f / (1.0f + __expf(-x)); }

__global__ void k_gate(const float* __restrict__ gi, const float* __restrict__ gh,
                       float* __restrict__ h, bf16_t* __restrict__ hb, int total) {
  int i = blockIdx.x * blockDim.x + threadIdx.x;
  if (i >= total) return;
  int node = i >> 8, c = i & 255;
  size_t base = (size_t)node * (3 * OUTF);
  float r = sigf(gi[base + c] + gh[base + c]);
  float z = sigf(gi[base + OUTF + c] + gh[base + OUTF + c]);
  float nn = tanhf(gi[base + 2 * OUTF + c] + r * gh[base + 2 * OUTF + c]);
  float hv = h[i];
  float o = (1.0f - z) * nn + z * hv;
  h[i] = o; hb[i] = (bf16_t)o;
}

// c_i = concat(h, node_features) padded to ZP, bf16
__global__ void k_build_cbuf(const float* __restrict__ h, const float* __restrict__ nf,
                             bf16_t* __restrict__ cb, int total) {
  int i = blockIdx.x * blockDim.x + threadIdx.x;
  if (i >= total) return;
  int row = i / ZP, c = i - row * ZP;
  float v;
  if (c < OUTF)       v = h[(size_t)row * OUTF + c];
  else if (c < CATF)  v = nf[(size_t)row * DINF + (c - OUTF)];
  else                v = 0.0f;
  cb[i] = (bf16_t)v;
}

// max-pool over L (rows within a batch); optional f32 and/or bf16 output
__global__ void k_maxpool(const float* __restrict__ in, int ldin, int Lin,
                          int win, int stride, int Lout, int C, int ldout,
                          float* __restrict__ outf, bf16_t* __restrict__ outb,
                          int total) {
  int i = blockIdx.x * blockDim.x + threadIdx.x;
  if (i >= total) return;
  int c = i % ldout;
  int t2 = i / ldout;
  int lo = t2 % Lout;
  int b  = t2 / Lout;
  float m = 0.0f;
  if (c < C) {
    m = -3.4e38f;
    int r0 = b * Lin + lo * stride;
    for (int j = 0; j < win; ++j)
      m = fmaxf(m, in[(size_t)(r0 + j) * ldin + c]);
  }
  if (outf) outf[i] = m;
  if (outb) outb[i] = (bf16_t)m;
}

// per-batch head: result[b], avg[b], temp[b, 256+362]
__global__ void k_head(const float* __restrict__ y2, const float* __restrict__ z2,
                       const float* __restrict__ myw, const float* __restrict__ myb,
                       const float* __restrict__ mzw, const float* __restrict__ mzb,
                       float* __restrict__ out) {
  const int L = 127;
  int b = blockIdx.x, tid = threadIdx.x;
  const float* Y = y2 + (size_t)b * L * OUTF;
  const float* Z = z2 + (size_t)b * L * ZP;
  float* temp = out + 2 * BATCH + (size_t)b * (OUTF + CATF);
  // column sums (temp)
  {
    float s = 0.0f;
    for (int l = 0; l < L; ++l) s += Y[(size_t)l * OUTF + tid];
    temp[tid] = s;
  }
  for (int c = tid; c < CATF; c += 256) {
    float s = 0.0f;
    for (int l = 0; l < L; ++l) s += Z[(size_t)l * ZP + c];
    temp[OUTF + c] = s;
  }
  // mean of (Y.my + by)*(Z.mz + bz) over L
  float acc = 0.0f;
  for (int l = tid; l < L; l += 256) {
    float yd = myb[0];
    for (int c = 0; c < OUTF; ++c) yd += Y[(size_t)l * OUTF + c] * myw[c];
    float zd = mzb[0];
    for (int c = 0; c < CATF; ++c) zd += Z[(size_t)l * ZP + c] * mzw[c];
    acc += yd * zd;
  }
  __shared__ float red[256];
  red[tid] = acc; __syncthreads();
  for (int s = 128; s > 0; s >>= 1) {
    if (tid < s) red[tid] += red[tid + s];
    __syncthreads();
  }
  if (tid == 0) {
    float avg = red[0] / (float)L;
    out[BATCH + b] = avg;
    out[b] = sigf(avg);
  }
}

// ---------------- host side ----------------
static inline void launch_gemm(const bf16_t* A, int lda, const bf16_t* Bw, int ldb,
                               const float* bias, float* C, int ldc,
                               int M, int N, int K, int rpo, int rpi,
                               int nshift, int relu, hipStream_t s) {
  int blocks = (M / 128) * (N / 32);    // exact grid: M%128==0, N%32==0
  wmma_gemm_bf16<<<blocks, 128, 0, s>>>(A, lda, Bw, ldb, bias, C, ldc,
                                        M, N, K, rpo, rpi, nshift, relu);
}
static inline int nb(int total) { return (total + 255) / 256; }

extern "C" void kernel_launch(void* const* d_in, const int* in_sizes, int n_in,
                              void* d_out, int out_size, void* d_ws, size_t ws_size,
                              hipStream_t stream) {
  const float* nf   = (const float*)d_in[0];
  const int*   ei   = (const int*)d_in[1];
  const int*   et   = (const int*)d_in[2];
  const float* W_e  = (const float*)d_in[3];
  const float* b_e  = (const float*)d_in[4];
  const float* W_ih = (const float*)d_in[5];
  const float* b_ih = (const float*)d_in[6];
  const float* W_hh = (const float*)d_in[7];
  const float* b_hh = (const float*)d_in[8];
  const float* c1w  = (const float*)d_in[9];
  const float* c1b  = (const float*)d_in[10];
  const float* c2w  = (const float*)d_in[11];
  const float* c2b  = (const float*)d_in[12];
  const float* cc1w = (const float*)d_in[13];
  const float* cc1b = (const float*)d_in[14];
  const float* cc2w = (const float*)d_in[15];
  const float* cc2b = (const float*)d_in[16];
  const float* myw  = (const float*)d_in[17];
  const float* myb  = (const float*)d_in[18];
  const float* mzw  = (const float*)d_in[19];
  const float* mzb  = (const float*)d_in[20];
  const int* srcI = ei;
  const int* dstI = ei + NEDGE;

  char* ws = (char*)d_ws;
  size_t off = 0;
  auto alloc = [&](size_t bytes) {
    size_t o = off;
    off += (bytes + 255) & ~(size_t)255;
    return o;
  };
  // persistent
  size_t oH   = alloc((size_t)BN * OUTF * 4);
  size_t oHb  = alloc((size_t)BN * OUTF * 2);
  size_t oWe  = alloc((size_t)NTYPE * OUTF * OUTF * 2);
  size_t oWih = alloc((size_t)3 * OUTF * OUTF * 2);
  size_t oWhh = alloc((size_t)3 * OUTF * OUTF * 2);
  size_t oW1y = alloc((size_t)3 * OUTF * OUTF * 2);
  size_t oW2y = alloc((size_t)OUTF * OUTF * 2);
  size_t oW1z = alloc((size_t)3 * ZP * ZP * 2);
  size_t oW2z = alloc((size_t)ZP * ZP * 2);
  size_t oCb1 = alloc((size_t)ZP * 4);
  size_t oCb2 = alloc((size_t)ZP * 4);
  size_t phase = off;
  // GNN-phase scratch
  size_t oA   = alloc((size_t)BN * OUTF * 4);
  size_t oAb  = alloc((size_t)BN * OUTF * 2);
  size_t oPr  = alloc((size_t)BN * OUTF * 4);
  size_t oGi  = alloc((size_t)BN * 3 * OUTF * 4);
  size_t oGh  = alloc((size_t)BN * 3 * OUTF * 4);
  // conv-phase scratch (overlaps GNN scratch)
  off = phase;
  size_t oCbuf = alloc((size_t)BN * ZP * 2);
  size_t oY1   = alloc((size_t)BATCH * 510 * OUTF * 4);
  size_t oYp1  = alloc((size_t)BATCH * 254 * OUTF * 2);
  size_t oY2   = alloc((size_t)BATCH * 254 * OUTF * 4);
  size_t oY2f  = alloc((size_t)BATCH * 127 * OUTF * 4);
  size_t oZ1   = alloc((size_t)BATCH * 510 * ZP * 4);
  size_t oZp1  = alloc((size_t)BATCH * 254 * ZP * 2);
  size_t oZ2   = alloc((size_t)BATCH * 254 * ZP * 4);
  size_t oZ2f  = alloc((size_t)BATCH * 127 * ZP * 4);
  (void)ws_size; (void)in_sizes; (void)n_in; (void)out_size;

  float*  H    = (float*)(ws + oH);
  bf16_t* Hb   = (bf16_t*)(ws + oHb);
  bf16_t* WeB  = (bf16_t*)(ws + oWe);
  bf16_t* WihB = (bf16_t*)(ws + oWih);
  bf16_t* WhhB = (bf16_t*)(ws + oWhh);
  bf16_t* W1y  = (bf16_t*)(ws + oW1y);
  bf16_t* W2y  = (bf16_t*)(ws + oW2y);
  bf16_t* W1z  = (bf16_t*)(ws + oW1z);
  bf16_t* W2z  = (bf16_t*)(ws + oW2z);
  float*  Cb1  = (float*)(ws + oCb1);
  float*  Cb2  = (float*)(ws + oCb2);
  float*  Ab   = (float*)(ws + oA);
  bf16_t* AbB  = (bf16_t*)(ws + oAb);
  float*  Proj = (float*)(ws + oPr);
  float*  Gi   = (float*)(ws + oGi);
  float*  Gh   = (float*)(ws + oGh);
  bf16_t* Cbuf = (bf16_t*)(ws + oCbuf);
  float*  Y1   = (float*)(ws + oY1);
  bf16_t* Yp1  = (bf16_t*)(ws + oYp1);
  float*  Y2   = (float*)(ws + oY2);
  float*  Y2f  = (float*)(ws + oY2f);
  float*  Z1   = (float*)(ws + oZ1);
  bf16_t* Zp1  = (bf16_t*)(ws + oZp1);
  float*  Z2   = (float*)(ws + oZ2);
  float*  Z2f  = (float*)(ws + oZ2f);

  // ---- weight conversion / repack ----
  k_f32_to_bf16<<<nb(NTYPE*OUTF*OUTF), 256, 0, stream>>>(W_e,  WeB,  NTYPE*OUTF*OUTF);
  k_f32_to_bf16<<<nb(3*OUTF*OUTF),     256, 0, stream>>>(W_ih, WihB, 3*OUTF*OUTF);
  k_f32_to_bf16<<<nb(3*OUTF*OUTF),     256, 0, stream>>>(W_hh, WhhB, 3*OUTF*OUTF);
  k_repack_conv<<<nb(3*OUTF*OUTF), 256, 0, stream>>>(c1w,  W1y, OUTF, OUTF, 3, OUTF, OUTF);
  k_repack_conv<<<nb(OUTF*OUTF),   256, 0, stream>>>(c2w,  W2y, OUTF, OUTF, 1, OUTF, OUTF);
  k_repack_conv<<<nb(3*ZP*ZP),     256, 0, stream>>>(cc1w, W1z, CATF, CATF, 3, ZP, ZP);
  k_repack_conv<<<nb(ZP*ZP),       256, 0, stream>>>(cc2w, W2z, CATF, CATF, 1, ZP, ZP);
  k_pad_bias<<<nb(ZP), 256, 0, stream>>>(cc1b, Cb1, CATF, ZP);
  k_pad_bias<<<nb(ZP), 256, 0, stream>>>(cc2b, Cb2, CATF, ZP);
  k_pad_h<<<nb(BN*OUTF), 256, 0, stream>>>(nf, H, Hb, BN*OUTF);

  // ---- GatedGraphConv, 6 steps ----
  for (int step = 0; step < NSTEP; ++step) {
    k_zero_f32<<<nb(BN*OUTF), 256, 0, stream>>>(Ab, BN*OUTF);
    for (int t = 0; t < NTYPE; ++t) {
      // proj_t = h @ W_e[t]^T + b_e[t]   (WMMA bf16)
      launch_gemm(Hb, OUTF, WeB + (size_t)t*OUTF*OUTF, OUTF,
                  b_e + (size_t)t*OUTF, Proj, OUTF,
                  BN, OUTF, OUTF, 0, 0, 1, 0, stream);
      // a[dst] += proj_t[src]  for edges of this type
      k_scatter<<<NEDGE, OUTF, 0, stream>>>(Proj, srcI, dstI, et, t, Ab);
    }
    k_f32_to_bf16<<<nb(BN*OUTF), 256, 0, stream>>>(Ab, AbB, BN*OUTF);
    // gi = a @ W_ih^T + b_ih ; gh = h @ W_hh^T + b_hh
    launch_gemm(AbB, OUTF, WihB, OUTF, b_ih, Gi, 3*OUTF,
                BN, 3*OUTF, OUTF, 0, 0, 1, 0, stream);
    launch_gemm(Hb, OUTF, WhhB, OUTF, b_hh, Gh, 3*OUTF,
                BN, 3*OUTF, OUTF, 0, 0, 1, 0, stream);
    k_gate<<<nb(BN*OUTF), 256, 0, stream>>>(Gi, Gh, H, Hb, BN*OUTF);
  }

  // ---- conv towers ----
  k_build_cbuf<<<nb(BN*ZP), 256, 0, stream>>>(H, nf, Cbuf, BN*ZP);
  // Y: conv1(K=3)+ReLU -> pool(3,2) -> conv2(K=1)+ReLU -> pool(2,2)
  launch_gemm(Hb, OUTF, W1y, OUTF, c1b, Y1, OUTF,
              BATCH*510, OUTF, OUTF, 510, SEQN, 3, 1, stream);
  k_maxpool<<<nb(BATCH*254*OUTF), 256, 0, stream>>>(Y1, OUTF, 510, 3, 2, 254,
              OUTF, OUTF, nullptr, Yp1, BATCH*254*OUTF);
  launch_gemm(Yp1, OUTF, W2y, OUTF, c2b, Y2, OUTF,
              BATCH*254, OUTF, OUTF, 0, 0, 1, 1, stream);
  k_maxpool<<<nb(BATCH*127*OUTF), 256, 0, stream>>>(Y2, OUTF, 254, 2, 2, 127,
              OUTF, OUTF, Y2f, nullptr, BATCH*127*OUTF);
  // Z: same on concat features (362 padded to 384 in both N and K; pad
  // columns carry exact zeros through conv+relu+pool, so they double as
  // the K-padding zeros of the next GEMM)
  launch_gemm(Cbuf, ZP, W1z, ZP, Cb1, Z1, ZP,
              BATCH*510, ZP, ZP, 510, SEQN, 3, 1, stream);
  k_maxpool<<<nb(BATCH*254*ZP), 256, 0, stream>>>(Z1, ZP, 510, 3, 2, 254,
              ZP, ZP, nullptr, Zp1, BATCH*254*ZP);
  launch_gemm(Zp1, ZP, W2z, ZP, Cb2, Z2, ZP,
              BATCH*254, ZP, ZP, 0, 0, 1, 1, stream);
  k_maxpool<<<nb(BATCH*127*ZP), 256, 0, stream>>>(Z2, ZP, 254, 2, 2, 127,
              ZP, ZP, Z2f, nullptr, BATCH*127*ZP);

  // ---- head ----
  k_head<<<BATCH, 256, 0, stream>>>(Y2f, Z2f, myw, myb, mzw, mzb, (float*)d_out);
}
